// CrossAttentionWithEmbedding_34282428956867
// MI455X (gfx1250) — matrix operand order repaired and checked by reading, not compiled, not run on hardware
//
#include <hip/hip_runtime.h>

typedef __attribute__((ext_vector_type(16))) _Float16 v16h;
typedef __attribute__((ext_vector_type(8)))  _Float16 v8h;
typedef __attribute__((ext_vector_type(8)))  float    v8f;
typedef __attribute__((ext_vector_type(4)))  unsigned v4u;
typedef __attribute__((ext_vector_type(8)))  unsigned v8u;

#define N_TOK 6400
#define CDIM  128

// ---------------- workspace layout (bytes) ----------------
#define OFF_XT    0u                          // 6400*128 f16
#define OFF_WH    1638400u                    // 128*128 f16
#define OFF_YPRE  1671168u                    // 6400*128 f32
#define OFF_QH    4947968u                    // 6400*128 f16
#define OFF_KH    6586368u
#define OFF_VH    8224768u
#define OFF_OH    9863168u
#define OFF_HH    11501568u
#define OFF_QRS   13139968u                   // 6400 f32
#define OFF_MEAN  13165568u                   // 128 f32
#define OFF_RSTD  13166080u                   // 128 f32

// ---------------- LDS / DMA helpers ----------------
__device__ __forceinline__ unsigned lds_off_u32(const void* p) {
    return (unsigned)(unsigned long long)(__attribute__((address_space(3))) const char*)p;
}

// Async copy of one 32x128 f16 tile (8 KB) from global to LDS, 128 threads,
// 4 x b128 per thread.  Tracked by ASYNCcnt.
__device__ __forceinline__ void async_tile_copy(const _Float16* __restrict__ gsrc,
                                                _Float16* ldst, int tid) {
    #pragma unroll
    for (int i = 0; i < 4; ++i) {
        int c = tid + i * 128;                      // chunk of 8 halves
        unsigned loff = lds_off_u32(ldst + (size_t)c * 8);
        unsigned long long ga = (unsigned long long)(gsrc + (size_t)c * 8);
        asm volatile("global_load_async_to_lds_b128 %0, %1, off"
                     :: "v"(loff), "v"(ga) : "memory");
    }
}

__device__ __forceinline__ void wait_async0() {
    asm volatile("s_wait_asynccnt 0" ::: "memory");
}

// Tensor Data Mover: one instruction moves a 32-row x 128-col f16 tile
// (row-major, row stride = 128 elements) from global to LDS.  2D tensor ->
// 2-SGPR-group D# (VADDR2/3 = NULL).  Tracked by TENSORcnt.
__device__ __forceinline__ void tdm_load_tile_2d(unsigned lds_addr,
                                                 const _Float16* gptr) {
    unsigned long long ga = (unsigned long long)gptr;
    // D# group 0: count=1 (valid), lds_addr, 57-bit global addr, type=2
    v4u g0 = { 1u,
               lds_addr,
               (unsigned)ga,
               (unsigned)((ga >> 32) & 0x01FFFFFFu) | (2u << 30) };
    // D# group 1: data_size=1 (2-byte elems); tensor_dim0=128; tensor_dim1=6400;
    // tile_dim0=128; tile_dim1=32; tensor_dim0_stride=128.
    v8u g1 = { 0x00010000u,
               ((unsigned)CDIM) << 16,
               ((unsigned)N_TOK) << 16,
               ((unsigned)CDIM) << 16,
               32u,
               (unsigned)CDIM,
               0u, 0u };
    asm volatile("tensor_load_to_lds %0, %1" :: "s"(g0), "s"(g1) : "memory");
}

// ---------------- WMMA fragment helpers ----------------
// A-matrix 16x32 f16 layout: lane<16 -> row=lane, K 0..15;
// lane>=16 -> row=lane-16, K 16..31.  Two contiguous 8-half runs.
__device__ __forceinline__ v16h load_fragA(const _Float16* base, int ld, int lane) {
    int row  = lane & 15;
    int koff = (lane & 16) ? 16 : 0;
    const _Float16* p = base + row * ld + koff;
    v8h lo = *(const v8h*)(p);
    v8h hi = *(const v8h*)(p + 8);
    return __builtin_shufflevector(lo, hi, 0,1,2,3,4,5,6,7,8,9,10,11,12,13,14,15);
}

// B-matrix 32x16 f16: lane<16 -> col=lane, K 0..15; lane>=16 -> col=lane-16, K 16..31.
__device__ __forceinline__ v16h load_fragB(const _Float16* base, int ld, int lane) {
    int col  = lane & 15;
    int koff = (lane & 16) ? 16 : 0;
    const _Float16* p = base + col * ld + koff;
    v8h lo = *(const v8h*)(p);
    v8h hi = *(const v8h*)(p + 8);
    return __builtin_shufflevector(lo, hi, 0,1,2,3,4,5,6,7,8,9,10,11,12,13,14,15);
}

// B-matrix fragment from a row-major [K][N] tile: per-lane strided gather.
__device__ __forceinline__ v16h load_fragB_strided(const _Float16* base, int ld, int lane) {
    int col  = lane & 15;
    int koff = (lane & 16) ? 16 : 0;
    const _Float16* p = base + (size_t)koff * ld + col;
    v16h r;
    #pragma unroll
    for (int i = 0; i < 16; ++i) r[i] = p[(size_t)i * ld];
    return r;
}

__device__ __forceinline__ v8f wmma_f16(v16h a, v16h b, v8f c) {
    return __builtin_amdgcn_wmma_f32_16x16x32_f16(false, a, false, b, (short)0, c, false, false);
}

// ---------------- small elementwise kernels ----------------
__global__ void cvt_f16_k(const float* __restrict__ w, _Float16* __restrict__ wh) {
    int i = blockIdx.x * blockDim.x + threadIdx.x;   // 16384 threads
    wh[i] = (_Float16)w[i];
}

__global__ void transpose_cvt_k(const float* __restrict__ x, _Float16* __restrict__ xt) {
    int i = blockIdx.x * blockDim.x + threadIdx.x;   // N*C threads
    int n = i >> 7, c = i & 127;
    xt[i] = (_Float16)x[(size_t)c * N_TOK + n];
}

__global__ void __launch_bounds__(256) chan_stats_k(const float* __restrict__ Y,
                                                    float* __restrict__ mean,
                                                    float* __restrict__ rstd) {
    int o = blockIdx.x;
    float s = 0.f, sq = 0.f;
    for (int n = threadIdx.x; n < N_TOK; n += 256) {
        float v = Y[(size_t)n * CDIM + o];
        s += v; sq += v * v;
    }
    __shared__ float sm[256], sm2[256];
    sm[threadIdx.x] = s; sm2[threadIdx.x] = sq;
    __syncthreads();
    for (int k = 128; k > 0; k >>= 1) {
        if (threadIdx.x < (unsigned)k) {
            sm[threadIdx.x]  += sm[threadIdx.x + k];
            sm2[threadIdx.x] += sm2[threadIdx.x + k];
        }
        __syncthreads();
    }
    if (threadIdx.x == 0) {
        float mu  = sm[0] / (float)N_TOK;
        float var = sm2[0] / (float)N_TOK - mu * mu;
        mean[o] = mu;
        rstd[o] = rsqrtf(var + 1e-5f);
    }
}

__global__ void bn_relu_cvt_k(const float* __restrict__ Y,
                              const float* __restrict__ mean,
                              const float* __restrict__ rstd,
                              const float* __restrict__ g,
                              const float* __restrict__ beta,
                              _Float16* __restrict__ outh) {
    int i = blockIdx.x * blockDim.x + threadIdx.x;
    int o = i & (CDIM - 1);
    float v = (Y[i] - mean[o]) * rstd[o] * g[o] + beta[o];
    outh[i] = (_Float16)fmaxf(v, 0.f);
}

__global__ void row_sum_k(const _Float16* __restrict__ qh, float* __restrict__ qrs) {
    int n = blockIdx.x * blockDim.x + threadIdx.x;
    if (n >= N_TOK) return;
    float s = 0.f;
    #pragma unroll 8
    for (int c = 0; c < CDIM; ++c) s += (float)qh[(size_t)n * CDIM + c];
    qrs[n] = s;
}

// ---------------- generic GEMM: Y[n,o] = sum_c A[n,c]*W[o,c] (+bias) ----------------
__global__ void __launch_bounds__(128) gemm_awt_k(const _Float16* __restrict__ A,
                                                  const _Float16* __restrict__ W,
                                                  const float* __restrict__ bias,
                                                  float* __restrict__ Y,
                                                  int strideN, int strideO) {
    int lane = threadIdx.x & 31;
    int wave = threadIdx.x >> 5;
    int m0   = (blockIdx.x * 4 + wave) * 16;
    int hh   = (lane >> 4) & 1;
    int nc   = lane & 15;

    __builtin_prefetch(W + (size_t)threadIdx.x * 128, 0, 0);   // global_prefetch_b8

    v16h aF[4];
    #pragma unroll
    for (int kc = 0; kc < 4; ++kc)
        aF[kc] = load_fragA(A + (size_t)m0 * CDIM + kc * 32, CDIM, lane);

    #pragma unroll
    for (int oc = 0; oc < 8; ++oc) {
        v8f acc = {};
        #pragma unroll
        for (int kc = 0; kc < 4; ++kc) {
            v16h bF = load_fragB(W + (size_t)(oc * 16) * CDIM + kc * 32, CDIM, lane);
            acc = wmma_f16(aF[kc], bF, acc);
        }
        int o = oc * 16 + nc;
        float bv = bias ? bias[o] : 0.f;
        #pragma unroll
        for (int r = 0; r < 8; ++r) {
            int m = m0 + r + 8 * hh;
            Y[(size_t)m * strideN + (size_t)o * strideO] = acc[r] + bv;
        }
    }
}

// ---------------- flash attention with outer-product positional term ----------------
__global__ void __launch_bounds__(128) flash_attn_k(const _Float16* __restrict__ qh,
                                                    const _Float16* __restrict__ kh,
                                                    const _Float16* __restrict__ vh,
                                                    const float* __restrict__ pos,
                                                    const float* __restrict__ qrs,
                                                    _Float16* __restrict__ oh) {
    __shared__ __align__(16) _Float16 sK[2][32 * CDIM];   // [buf][n_local][c]
    __shared__ __align__(16) _Float16 sV[2][32 * CDIM];   // [buf][n_local][c]
    __shared__ __align__(16) _Float16 sP[4 * 16 * 32];    // per-wave [m][j_local]

    int tid  = threadIdx.x;
    int lane = tid & 31;
    int wave = tid >> 5;
    int m0   = (blockIdx.x * 4 + wave) * 16;
    int hh   = (lane >> 4) & 1;
    int nc   = lane & 15;

    v16h aQ[4];
    #pragma unroll
    for (int kc = 0; kc < 4; ++kc)
        aQ[kc] = load_fragA(qh + (size_t)m0 * CDIM + kc * 32, CDIM, lane);

    float qr[8];
    #pragma unroll
    for (int r = 0; r < 8; ++r) qr[r] = qrs[m0 + r + 8 * hh];

    v8f O[8];
    #pragma unroll
    for (int oc = 0; oc < 8; ++oc) O[oc] = (v8f){};

    float mst[8], lst[8];
    #pragma unroll
    for (int r = 0; r < 8; ++r) { mst[r] = -3.0e38f; lst[r] = 0.f; }

    const float scale = 0.08838834764831845f;   // 1/sqrt(128)
    const float L2E   = 1.44269504088896f;
    _Float16* sPw = sP + wave * (16 * 32);

    // prologue: stage tile 0 (K via async-to-LDS, V via Tensor Data Mover)
    async_tile_copy(kh, sK[0], tid);
    if (tid < 32) {
        tdm_load_tile_2d(lds_off_u32(sV[0]), vh);
        __builtin_amdgcn_s_wait_tensorcnt(0);
    }
    wait_async0();
    __syncthreads();

    for (int j0 = 0; j0 < N_TOK; j0 += 32) {
        int cur = (j0 >> 5) & 1;
        // overlap: stage next tile into the other buffer while computing
        if (j0 + 32 < N_TOK) {
            async_tile_copy(kh + (size_t)(j0 + 32) * CDIM, sK[1 - cur], tid);
            if (tid < 32)
                tdm_load_tile_2d(lds_off_u32(sV[1 - cur]),
                                 vh + (size_t)(j0 + 32) * CDIM);
        }
        const _Float16* cK = sK[cur];
        const _Float16* cV = sV[cur];

        // S tile 16x32 = Q(16x128) @ K^T
        v8f s0 = {}, s1 = {};
        #pragma unroll
        for (int kc = 0; kc < 4; ++kc) {
            v16h b0 = load_fragB(cK + kc * 32, CDIM, lane);
            v16h b1 = load_fragB(cK + 16 * CDIM + kc * 32, CDIM, lane);
            s0 = wmma_f16(aQ[kc], b0, s0);
            s1 = wmma_f16(aQ[kc], b1, s1);
        }

        float p0v = pos[j0 + nc];
        float p1v = pos[j0 + 16 + nc];

        #pragma unroll
        for (int r = 0; r < 8; ++r) {
            float a0 = s0[r] * scale + qr[r] * p0v;
            float a1 = s1[r] * scale + qr[r] * p1v;
            float mx = fmaxf(a0, a1);
            mx = fmaxf(mx, __shfl_xor(mx, 1, 32));
            mx = fmaxf(mx, __shfl_xor(mx, 2, 32));
            mx = fmaxf(mx, __shfl_xor(mx, 4, 32));
            mx = fmaxf(mx, __shfl_xor(mx, 8, 32));
            float mnew = fmaxf(mst[r], mx);
            float corr = __builtin_exp2f((mst[r] - mnew) * L2E);
            float e0 = __builtin_exp2f((a0 - mnew) * L2E);
            float e1 = __builtin_exp2f((a1 - mnew) * L2E);
            float ts = e0 + e1;
            ts += __shfl_xor(ts, 1, 32);
            ts += __shfl_xor(ts, 2, 32);
            ts += __shfl_xor(ts, 4, 32);
            ts += __shfl_xor(ts, 8, 32);
            lst[r] = lst[r] * corr + ts;
            mst[r] = mnew;
            #pragma unroll
            for (int oc = 0; oc < 8; ++oc) O[oc][r] *= corr;
            int m = r + 8 * hh;
            sPw[m * 32 + nc]      = (_Float16)e0;
            sPw[m * 32 + 16 + nc] = (_Float16)e1;
        }

        // P written in C-layout, re-read in A-layout (same-wave LDS RAW)
        asm volatile("s_wait_dscnt 0" ::: "memory");
        v16h aP = load_fragA(sPw, 32, lane);
        #pragma unroll
        for (int oc = 0; oc < 8; ++oc) {
            v16h bV = load_fragB_strided(cV + oc * 16, CDIM, lane);
            O[oc] = wmma_f16(aP, bV, O[oc]);
        }

        // both DMA paths must land before anyone flips buffers
        wait_async0();
        if (tid < 32) __builtin_amdgcn_s_wait_tensorcnt(0);
        __syncthreads();
    }

    // normalize and store f16 [n,o]
    #pragma unroll
    for (int r = 0; r < 8; ++r) {
        float inv = 1.0f / lst[r];
        int m = m0 + r + 8 * hh;
        #pragma unroll
        for (int oc = 0; oc < 8; ++oc)
            oh[(size_t)m * CDIM + oc * 16 + nc] = (_Float16)(O[oc][r] * inv);
    }
}

// ---------------- host launch ----------------
extern "C" void kernel_launch(void* const* d_in, const int* in_sizes, int n_in,
                              void* d_out, int out_size, void* d_ws, size_t ws_size,
                              hipStream_t stream) {
    (void)in_sizes; (void)n_in; (void)out_size; (void)ws_size;
    const float* query = (const float*)d_in[0];
    const float* key_  = (const float*)d_in[1];
    const float* value = (const float*)d_in[2];
    const float* pos   = (const float*)d_in[3];
    const float* wq = (const float*)d_in[4];
    const float* gq = (const float*)d_in[6];
    const float* betaq = (const float*)d_in[7];
    const float* wk = (const float*)d_in[8];
    const float* gk = (const float*)d_in[10];
    const float* betak = (const float*)d_in[11];
    const float* wv = (const float*)d_in[12];
    const float* gv = (const float*)d_in[14];
    const float* betav = (const float*)d_in[15];
    const float* wo1 = (const float*)d_in[16];
    const float* go = (const float*)d_in[18];
    const float* betao = (const float*)d_in[19];
    const float* wo2 = (const float*)d_in[20];
    const float* bo2 = (const float*)d_in[21];

    char* ws = (char*)d_ws;
    _Float16* xT   = (_Float16*)(ws + OFF_XT);
    _Float16* wh   = (_Float16*)(ws + OFF_WH);
    float*    ypre = (float*)   (ws + OFF_YPRE);
    _Float16* qhp  = (_Float16*)(ws + OFF_QH);
    _Float16* khp  = (_Float16*)(ws + OFF_KH);
    _Float16* vhp  = (_Float16*)(ws + OFF_VH);
    _Float16* ohp  = (_Float16*)(ws + OFF_OH);
    _Float16* hhp  = (_Float16*)(ws + OFF_HH);
    float*    qrsp = (float*)   (ws + OFF_QRS);
    float*    meanb= (float*)   (ws + OFF_MEAN);
    float*    rstdb= (float*)   (ws + OFF_RSTD);

    struct Proj { const float* x; const float* w; const float* g; const float* b; _Float16* outh; };
    Proj P[3] = { {query, wq, gq, betaq, qhp},
                  {key_,  wk, gk, betak, khp},
                  {value, wv, gv, betav, vhp} };

    for (int p = 0; p < 3; ++p) {
        cvt_f16_k      <<<64,   256, 0, stream>>>(P[p].w, wh);
        transpose_cvt_k<<<3200, 256, 0, stream>>>(P[p].x, xT);
        gemm_awt_k     <<<100,  128, 0, stream>>>(xT, wh, nullptr, ypre, CDIM, 1);
        chan_stats_k   <<<128,  256, 0, stream>>>(ypre, meanb, rstdb);
        bn_relu_cvt_k  <<<3200, 256, 0, stream>>>(ypre, meanb, rstdb, P[p].g, P[p].b, P[p].outh);
    }
    row_sum_k   <<<25,  256, 0, stream>>>(qhp, qrsp);
    flash_attn_k<<<100, 128, 0, stream>>>(qhp, khp, vhp, pos, qrsp, ohp);

    cvt_f16_k    <<<64,   256, 0, stream>>>(wo1, wh);
    gemm_awt_k   <<<100,  128, 0, stream>>>(ohp, wh, nullptr, ypre, CDIM, 1);
    chan_stats_k <<<128,  256, 0, stream>>>(ypre, meanb, rstdb);
    bn_relu_cvt_k<<<3200, 256, 0, stream>>>(ypre, meanb, rstdb, go, betao, hhp);

    cvt_f16_k  <<<64,  256, 0, stream>>>(wo2, wh);
    gemm_awt_k <<<100, 128, 0, stream>>>(hhp, wh, bo2, (float*)d_out, 1, N_TOK);
}
